// ComplexMultiheadAttention_28965259444418
// MI455X (gfx1250) — matrix operand order repaired and checked by reading, not compiled, not run on hardware
//
#include <hip/hip_runtime.h>

typedef _Float16 v16h __attribute__((ext_vector_type(16)));
typedef _Float16 v8h  __attribute__((ext_vector_type(8)));
typedef float    v8f  __attribute__((ext_vector_type(8)));

#define EDIM 1024
#define SEQ  1024
#define NB   4
#define NH   16
#define HD   64

__device__ __forceinline__ v16h cat8(v8h lo, v8h hi) {
  return __builtin_shufflevector(lo, hi, 0,1,2,3,4,5,6,7,8,9,10,11,12,13,14,15);
}

__device__ __forceinline__ v8f wmma16(v16h a, v16h b, v8f c) {
  // v_wmma_f32_16x16x32_f16
  return __builtin_amdgcn_wmma_f32_16x16x32_f16(false, a, false, b, (short)0, c, false, false);
}

__device__ __forceinline__ v8f zero8() {
  v8f z;
#pragma unroll
  for (int i = 0; i < 8; ++i) z[i] = 0.0f;
  return z;
}

// ---------------------------------------------------------------------------
// Generic projection GEMM: C[M=4096 x N=1024] = A[4096 x 1024] * W[1024 x 1024]^T + bias
// MODE 0: f16 out, head layout      [b,h,s,d]
// MODE 1: f16 out, transposed head  [b,h,d,s]   (for V, so PV B-operand is K-contiguous)
// MODE 2: f32 out, row-major        [m,n]       (final projections -> d_out)
// AF16 : A source is f16 (1) or f32 (0)
// ---------------------------------------------------------------------------
template<int MODE, int AF16>
__global__ __launch_bounds__(256) void gemm_kernel(
    const void* __restrict__ Av, const float* __restrict__ W,
    const float* __restrict__ bias, _Float16* __restrict__ outH,
    float* __restrict__ outF)
{
  __shared__ __align__(16) _Float16 ldsA[128 * 40];
  __shared__ __align__(16) _Float16 ldsB[128 * 40];

  const int t     = threadIdx.x;
  const int mTile = blockIdx.x;          // 0..31
  const int nTile = blockIdx.y;          // 0..7
  const int w     = t >> 5;
  const int lane  = t & 31;
  const int l16   = lane & 15;
  const int h2    = lane >> 4;
  const int rBase = (w >> 2) * 64;       // 2 row groups of 64
  const int cBase = (w & 3) * 32;        // 4 col groups of 32
  const int sRow  = t >> 1;              // staging: 2 threads per row
  const int sCol  = (t & 1) * 16;

  const float*    A32 = (const float*)Av;
  const _Float16* A16 = (const _Float16*)Av;
  const long aOff = (long)(mTile * 128 + sRow) * EDIM + sCol;
  const long bOff = (long)(nTile * 128 + sRow) * EDIM + sCol;

  v8f acc[4][2];
#pragma unroll
  for (int i = 0; i < 4; ++i)
#pragma unroll
    for (int j = 0; j < 2; ++j) acc[i][j] = zero8();

  for (int k0 = 0; k0 < EDIM; k0 += 32) {
    __syncthreads();
    // ---- stage A tile (convert to f16 if needed) ----
    v8h a0, a1;
    if (AF16) {
      a0 = *(const v8h*)(A16 + aOff + k0);
      a1 = *(const v8h*)(A16 + aOff + k0 + 8);
    } else {
      const float* ap = A32 + aOff + k0;
#pragma unroll
      for (int i = 0; i < 8; ++i) { a0[i] = (_Float16)ap[i]; a1[i] = (_Float16)ap[8 + i]; }
    }
    *(v8h*)(ldsA + sRow * 40 + sCol)     = a0;
    *(v8h*)(ldsA + sRow * 40 + sCol + 8) = a1;
    // ---- stage B tile (weights, always f32 source) ----
    {
      const float* bp = W + bOff + k0;
      v8h b0, b1;
#pragma unroll
      for (int i = 0; i < 8; ++i) { b0[i] = (_Float16)bp[i]; b1[i] = (_Float16)bp[8 + i]; }
      *(v8h*)(ldsB + sRow * 40 + sCol)     = b0;
      *(v8h*)(ldsB + sRow * 40 + sCol + 8) = b1;
    }
    __syncthreads();

    // ---- fragments (contiguous k = 16*h2 + s in both A and B) ----
    v16h afr[4];
#pragma unroll
    for (int mt = 0; mt < 4; ++mt) {
      const _Float16* pa = ldsA + (rBase + mt * 16 + l16) * 40;
      afr[mt] = cat8(*(const v8h*)(pa + h2 * 16), *(const v8h*)(pa + h2 * 16 + 8));
    }
    v16h bfr[2];
#pragma unroll
    for (int nt = 0; nt < 2; ++nt) {
      const _Float16* pb = ldsB + (cBase + nt * 16 + l16) * 40;
      bfr[nt] = cat8(*(const v8h*)(pb + h2 * 16), *(const v8h*)(pb + h2 * 16 + 8));
    }
#pragma unroll
    for (int mt = 0; mt < 4; ++mt)
#pragma unroll
      for (int nt = 0; nt < 2; ++nt)
        acc[mt][nt] = wmma16(afr[mt], bfr[nt], acc[mt][nt]);
  }

  // ---- epilogue: bias add + layout-specific scatter ----
#pragma unroll
  for (int mt = 0; mt < 4; ++mt) {
#pragma unroll
    for (int nt = 0; nt < 2; ++nt) {
      const int n  = nTile * 128 + cBase + nt * 16 + l16;
      const float bv = bias[n];
#pragma unroll
      for (int j = 0; j < 8; ++j) {
        const int m   = mTile * 128 + rBase + mt * 16 + j + 8 * h2;
        const float v = acc[mt][nt][j] + bv;
        if (MODE == 2) {
          outF[(long)m * EDIM + n] = v;
        } else {
          const int bi = m >> 10, s = m & 1023;
          const int hh = n >> 6,  d = n & 63;
          long o;
          if (MODE == 0) o = (((long)(bi * NH + hh)) * SEQ + s) * HD + d;
          else           o = (((long)(bi * NH + hh)) * HD + d) * SEQ + s;
          outH[o] = (_Float16)v;
        }
      }
    }
  }
}

// ---------------------------------------------------------------------------
// Complex-magnitude attention. One workgroup per (b,h, 64-query block);
// each wave owns one 16-query WMMA tile. Max-free softmax (mag >= 0, small).
// ---------------------------------------------------------------------------
__global__ __launch_bounds__(128) void attn_kernel(
    const _Float16* __restrict__ qr, const _Float16* __restrict__ qi,
    const _Float16* __restrict__ kr, const _Float16* __restrict__ ki,
    const _Float16* __restrict__ vrT, const _Float16* __restrict__ viT,
    _Float16* __restrict__ outr, _Float16* __restrict__ outi)
{
  __shared__ __align__(16) _Float16 ldsP[4][16 * 64];  // wave-private P tiles

  const int bid  = blockIdx.x;
  const int bh   = bid >> 4;     // (b*16 + h), 0..63
  const int qblk = bid & 15;
  const int b    = bh >> 4, h = bh & 15;
  const int t    = threadIdx.x;
  const int w    = t >> 5;
  const int lane = t & 31, l16 = lane & 15, h2 = lane >> 4;
  const int qBase = qblk * 64 + w * 16;
  const long hd  = (long)bh * (SEQ * HD);

  // resident Q fragments (A layout), d split into two K=32 halves
  const _Float16* qrp = qr + hd + (long)(qBase + l16) * HD;
  const _Float16* qip = qi + hd + (long)(qBase + l16) * HD;
  v16h Qr0 = cat8(*(const v8h*)(qrp + h2 * 16),      *(const v8h*)(qrp + h2 * 16 + 8));
  v16h Qr1 = cat8(*(const v8h*)(qrp + 32 + h2 * 16), *(const v8h*)(qrp + 32 + h2 * 16 + 8));
  v16h Qi0 = cat8(*(const v8h*)(qip + h2 * 16),      *(const v8h*)(qip + h2 * 16 + 8));
  v16h Qi1 = cat8(*(const v8h*)(qip + 32 + h2 * 16), *(const v8h*)(qip + 32 + h2 * 16 + 8));
  v16h Qn0 = -Qr0, Qn1 = -Qr1;   // for Si = Qi*Kr^T + (-Qr)*Ki^T

  v8f aOr[4], aOi[4];
#pragma unroll
  for (int i = 0; i < 4; ++i) { aOr[i] = zero8(); aOi[i] = zero8(); }
  v8f rs = zero8();

  _Float16* myP = &ldsP[w][0];

  for (int kc = 0; kc < SEQ; kc += 64) {
    // ---- scores for 4 key tiles of 16 ----
#pragma unroll
    for (int kt = 0; kt < 4; ++kt) {
      const _Float16* kp  = kr + hd + (long)(kc + kt * 16 + l16) * HD;
      const _Float16* kip = ki + hd + (long)(kc + kt * 16 + l16) * HD;
      v16h Kr0 = cat8(*(const v8h*)(kp + h2 * 16),       *(const v8h*)(kp + h2 * 16 + 8));
      v16h Kr1 = cat8(*(const v8h*)(kp + 32 + h2 * 16),  *(const v8h*)(kp + 32 + h2 * 16 + 8));
      v16h Ki0 = cat8(*(const v8h*)(kip + h2 * 16),      *(const v8h*)(kip + h2 * 16 + 8));
      v16h Ki1 = cat8(*(const v8h*)(kip + 32 + h2 * 16), *(const v8h*)(kip + 32 + h2 * 16 + 8));

      v8f sr = zero8();
      sr = wmma16(Qr0, Kr0, sr); sr = wmma16(Qr1, Kr1, sr);
      sr = wmma16(Qi0, Ki0, sr); sr = wmma16(Qi1, Ki1, sr);
      v8f si = zero8();
      si = wmma16(Qi0, Kr0, si); si = wmma16(Qi1, Kr1, si);
      si = wmma16(Qn0, Ki0, si); si = wmma16(Qn1, Ki1, si);

#pragma unroll
      for (int j = 0; j < 8; ++j) {
        float mag = sqrtf(sr[j] * sr[j] + si[j] * si[j] + 1e-8f) * 0.125f;  // D^-0.5
        float p = expf(mag);
        rs[j] += p;
        myP[(j + 8 * h2) * 64 + kt * 16 + l16] = (_Float16)p;
      }
    }

    // ---- P tile back as A fragments (wave-private LDS, DS ops in-order) ----
    const _Float16* pp = myP + l16 * 64;
    v16h P0 = cat8(*(const v8h*)(pp + h2 * 16),      *(const v8h*)(pp + h2 * 16 + 8));
    v16h P1 = cat8(*(const v8h*)(pp + 32 + h2 * 16), *(const v8h*)(pp + 32 + h2 * 16 + 8));

    // ---- O += P * V  (V stored transposed [d][s] -> K-contiguous B frags) ----
#pragma unroll
    for (int dt = 0; dt < 4; ++dt) {
      const _Float16* vp  = vrT + hd + (long)(dt * 16 + l16) * SEQ + kc;
      const _Float16* vip = viT + hd + (long)(dt * 16 + l16) * SEQ + kc;
      v16h Vr0 = cat8(*(const v8h*)(vp + h2 * 16),       *(const v8h*)(vp + h2 * 16 + 8));
      v16h Vr1 = cat8(*(const v8h*)(vp + 32 + h2 * 16),  *(const v8h*)(vp + 32 + h2 * 16 + 8));
      v16h Vi0 = cat8(*(const v8h*)(vip + h2 * 16),      *(const v8h*)(vip + h2 * 16 + 8));
      v16h Vi1 = cat8(*(const v8h*)(vip + 32 + h2 * 16), *(const v8h*)(vip + 32 + h2 * 16 + 8));
      aOr[dt] = wmma16(P0, Vr0, aOr[dt]); aOr[dt] = wmma16(P1, Vr1, aOr[dt]);
      aOi[dt] = wmma16(P0, Vi0, aOi[dt]); aOi[dt] = wmma16(P1, Vi1, aOi[dt]);
    }
  }

  // ---- row-sum reduction over the 16 lanes holding one row, then normalize ----
#pragma unroll
  for (int j = 0; j < 8; ++j) {
    float v = rs[j];
    v += __shfl_xor(v, 1, 16);
    v += __shfl_xor(v, 2, 16);
    v += __shfl_xor(v, 4, 16);
    v += __shfl_xor(v, 8, 16);
    rs[j] = 1.0f / v;
  }

#pragma unroll
  for (int dt = 0; dt < 4; ++dt) {
#pragma unroll
    for (int j = 0; j < 8; ++j) {
      const int m   = qBase + j + 8 * h2;          // query row within S
      const int col = h * HD + dt * 16 + l16;      // column within E
      const long o  = ((long)(b * SEQ + m)) * EDIM + col;
      outr[o] = (_Float16)(aOr[dt][j] * rs[j]);
      outi[o] = (_Float16)(aOi[dt][j] * rs[j]);
    }
  }
}

// ---------------------------------------------------------------------------
extern "C" void kernel_launch(void* const* d_in, const int* in_sizes, int n_in,
                              void* d_out, int out_size, void* d_ws, size_t ws_size,
                              hipStream_t stream)
{
  const float* query_real = (const float*)d_in[0];
  const float* query_imag = (const float*)d_in[1];
  const float* key_real   = (const float*)d_in[2];
  const float* key_imag   = (const float*)d_in[3];
  const float* value_real = (const float*)d_in[4];
  const float* value_imag = (const float*)d_in[5];
  const float* Wq_r = (const float*)d_in[6];  const float* bq_r = (const float*)d_in[7];
  const float* Wq_i = (const float*)d_in[8];  const float* bq_i = (const float*)d_in[9];
  const float* Wk_r = (const float*)d_in[10]; const float* bk_r = (const float*)d_in[11];
  const float* Wk_i = (const float*)d_in[12]; const float* bk_i = (const float*)d_in[13];
  const float* Wv_r = (const float*)d_in[14]; const float* bv_r = (const float*)d_in[15];
  const float* Wv_i = (const float*)d_in[16]; const float* bv_i = (const float*)d_in[17];
  const float* Wo_r = (const float*)d_in[18]; const float* bo_r = (const float*)d_in[19];
  const float* Wo_i = (const float*)d_in[20]; const float* bo_i = (const float*)d_in[21];

  const size_t NE = (size_t)NB * SEQ * EDIM;   // 4M elements
  _Float16* ws   = (_Float16*)d_ws;            // 8 * NE * 2B = 64 MB scratch
  _Float16* qr   = ws + 0 * NE;
  _Float16* qi   = ws + 1 * NE;
  _Float16* kr   = ws + 2 * NE;
  _Float16* ki   = ws + 3 * NE;
  _Float16* vrT  = ws + 4 * NE;
  _Float16* viT  = ws + 5 * NE;
  _Float16* or16 = ws + 6 * NE;
  _Float16* oi16 = ws + 7 * NE;
  float* outF = (float*)d_out;

  dim3 g(32, 8), blk(256);
  gemm_kernel<0, 0><<<g, blk, 0, stream>>>(query_real, Wq_r, bq_r, qr,  nullptr);
  gemm_kernel<0, 0><<<g, blk, 0, stream>>>(query_imag, Wq_i, bq_i, qi,  nullptr);
  gemm_kernel<0, 0><<<g, blk, 0, stream>>>(key_real,   Wk_r, bk_r, kr,  nullptr);
  gemm_kernel<0, 0><<<g, blk, 0, stream>>>(key_imag,   Wk_i, bk_i, ki,  nullptr);
  gemm_kernel<1, 0><<<g, blk, 0, stream>>>(value_real, Wv_r, bv_r, vrT, nullptr);
  gemm_kernel<1, 0><<<g, blk, 0, stream>>>(value_imag, Wv_i, bv_i, viT, nullptr);

  attn_kernel<<<dim3(NB * NH * (SEQ / 64)), dim3(128), 0, stream>>>(
      qr, qi, kr, ki, vrT, viT, or16, oi16);

  gemm_kernel<2, 1><<<g, blk, 0, stream>>>(or16, Wo_r, bo_r, nullptr, outF);
  gemm_kernel<2, 1><<<g, blk, 0, stream>>>(oi16, Wo_i, bo_i, nullptr, outF + NE);
}
